// CLIPEncoderLayer_13924283974220
// MI455X (gfx1250) — matrix operand order repaired and checked by reading, not compiled, not run on hardware
//
#include <hip/hip_runtime.h>
#include <hip/hip_bf16.h>

// ---------------------------------------------------------------------------
// CLIP encoder layer on gfx1250 (MI455X), bf16 WMMA with f32 accumulate.
// GEMMs use double-buffered GLOBAL_LOAD_ASYNC_TO_LDS_B128 staging (ASYNCcnt).
// B=4, S=2048, D=1024, H=16, HD=64.
// ---------------------------------------------------------------------------

#define Bv   4
#define Sv   2048
#define Dv   1024
#define Hh   16
#define HDv  64
#define ROWS (Bv * Sv)          // 8192
#define DFF  (4 * Dv)           // 4096

typedef __attribute__((ext_vector_type(16))) __bf16 v16bf;
typedef __attribute__((ext_vector_type(8)))  float  v8f;

union FragU {
    uint4          q[2];
    v16bf          v;
    unsigned short h[16];
};

__device__ __forceinline__ unsigned short f2bf(float x) {
    unsigned u = __float_as_uint(x);
    unsigned r = ((u >> 16) & 1u) + 0x7FFFu;     // round-to-nearest-even
    return (unsigned short)((u + r) >> 16);
}

// A-matrix 16x32 bf16 fragment (ISA 7.12.2): lane<16 -> row, elems 0..7 = K k0+hi*8..,
// elems 8..15 = K k0+16+hi*8.. ; base is row-major [rows, ld].
__device__ __forceinline__ v16bf load_a(const unsigned short* __restrict__ base,
                                        int ld, int row0, int k0) {
    int lane = threadIdx.x & 31;
    const unsigned short* p =
        base + (long long)(row0 + (lane & 15)) * ld + k0 + ((lane >> 4) << 3);
    FragU f;
    f.q[0] = *(const uint4*)(p);
    f.q[1] = *(const uint4*)(p + 16);
    return f.v;
}

// B-matrix 32x16 bf16 fragment: lane -> col, elems 0..7 = K k0+hi*8..,
// elems 8..15 = K k0+16+hi*8.. . base is row-major [cols, ld] (i.e. B^T storage => B = base^T).
__device__ __forceinline__ v16bf load_b(const unsigned short* __restrict__ base,
                                        int ld, int col0, int k0) {
    int lane = threadIdx.x & 31;
    const unsigned short* p =
        base + (long long)(col0 + (lane & 15)) * ld + k0 + ((lane >> 4) << 3);
    FragU f;
    f.q[0] = *(const uint4*)(p);
    f.q[1] = *(const uint4*)(p + 16);
    return f.v;
}

// LDS-resident variants (tile leading dim = 32 elements), lowered to ds_load_b128.
__device__ __forceinline__ v16bf load_a_lds(const unsigned short* base, int row0) {
    int lane = threadIdx.x & 31;
    const unsigned short* p = base + (row0 + (lane & 15)) * 32 + ((lane >> 4) << 3);
    FragU f;
    f.q[0] = *(const uint4*)(p);
    f.q[1] = *(const uint4*)(p + 16);
    return f.v;
}
__device__ __forceinline__ v16bf load_b_lds(const unsigned short* base, int col0) {
    int lane = threadIdx.x & 31;
    const unsigned short* p = base + (col0 + (lane & 15)) * 32 + ((lane >> 4) << 3);
    FragU f;
    f.q[0] = *(const uint4*)(p);
    f.q[1] = *(const uint4*)(p + 16);
    return f.v;
}

#define WMMA_BF16(a, b, c) \
    __builtin_amdgcn_wmma_f32_16x16x32_bf16(false, (a), false, (b), (short)0, (c), false, false)

// Async copy: 16 bytes per lane, global -> LDS, tracked by ASYNCcnt.
__device__ __forceinline__ void async_cp_b128(unsigned lds_off, const void* gptr) {
    unsigned long long ga = (unsigned long long)(uintptr_t)gptr;
    asm volatile("global_load_async_to_lds_b128 %0, %1, off"
                 :: "v"(lds_off), "v"(ga) : "memory");
}
__device__ __forceinline__ void wait_async(int n) {
    if (n == 0) asm volatile("s_wait_asynccnt 0" ::: "memory");
    else        asm volatile("s_wait_asynccnt 3" ::: "memory");
}

// ---------------------------------------------------------------------------
// Weight fp32 [K,N] -> bf16 transposed [N,K]
// ---------------------------------------------------------------------------
__global__ __launch_bounds__(256) void wt_convert(const float* __restrict__ W,
                                                  unsigned short* __restrict__ Wt,
                                                  int K, int N) {
    long long t = (long long)blockIdx.x * 256 + threadIdx.x;
    if (t >= (long long)K * N) return;
    int k = (int)(t / N);
    int n = (int)(t % N);
    Wt[(long long)n * K + k] = f2bf(W[t]);
}

// ---------------------------------------------------------------------------
// LayerNorm: fp32 [rows, 1024] -> bf16 [rows, 1024]; one block (256 thr) per row
// ---------------------------------------------------------------------------
__global__ __launch_bounds__(256) void ln_kernel(const float* __restrict__ x,
                                                 const float* __restrict__ g,
                                                 const float* __restrict__ be,
                                                 unsigned short* __restrict__ out) {
    __shared__ float red[256];
    long long base = (long long)blockIdx.x * Dv;
    int t = threadIdx.x;
    float4 v = ((const float4*)(x + base))[t];
    red[t] = v.x + v.y + v.z + v.w;
    __syncthreads();
    for (int off = 128; off; off >>= 1) {
        if (t < off) red[t] += red[t + off];
        __syncthreads();
    }
    float mu = red[0] * (1.f / Dv);
    __syncthreads();
    float dx = v.x - mu, dy = v.y - mu, dz = v.z - mu, dw = v.w - mu;
    red[t] = dx * dx + dy * dy + dz * dz + dw * dw;
    __syncthreads();
    for (int off = 128; off; off >>= 1) {
        if (t < off) red[t] += red[t + off];
        __syncthreads();
    }
    float rs = rsqrtf(red[0] * (1.f / Dv) + 1e-5f);
    float4 gv = ((const float4*)g)[t];
    float4 bv = ((const float4*)be)[t];
    ushort4 o;
    o.x = f2bf(dx * rs * gv.x + bv.x);
    o.y = f2bf(dy * rs * gv.y + bv.y);
    o.z = f2bf(dz * rs * gv.z + bv.z);
    o.w = f2bf(dw * rs * gv.w + bv.w);
    ((ushort4*)(out + base))[t] = o;
}

// ---------------------------------------------------------------------------
// WMMA GEMM with double-buffered async-LDS staging.
// C[M,N] = A[M,K](bf16) * Bt[N,K]^T(bf16) + bias, templated epilogues.
// Block 256 thr = 8 waves; block tile 64(M) x 128(N); wave tile 16 x 64; K-step 32.
// Per K-step: A tile 64x32 (4KB), B tile 128x32 (8KB) staged via
// global_load_async_to_lds_b128 (3 issues/wave/stage), double buffered (24KB LDS).
// ---------------------------------------------------------------------------
enum { EPI_Q = 0, EPI_K = 1, EPI_VT = 2, EPI_RESF32 = 3, EPI_GELU = 4 };

template <int EPI>
__global__ __launch_bounds__(256) void gemm_k(const unsigned short* __restrict__ A,
                                              const unsigned short* __restrict__ Bt,
                                              const float* __restrict__ bias,
                                              const float* __restrict__ res,
                                              float* __restrict__ outF,
                                              unsigned short* __restrict__ outH,
                                              int M, int N, int K) {
    __shared__ unsigned short ldsA[2][64 * 32];
    __shared__ unsigned short ldsB[2][128 * 32];

    int tid = threadIdx.x;
    int wave = tid >> 5;
    int mblk = blockIdx.x * 64;
    int nblk = blockIdx.y * 128;
    int wm = (wave & 3) * 16;       // wave's M offset within LDS tile
    int wn = (wave >> 2) * 64;      // wave's N offset within LDS tile

    // Async-copy assignment: each thread moves one 16B chunk of A and two of B.
    int arow = tid >> 2;            // 0..63
    int ach  = (tid & 3) << 3;      // element offset 0/8/16/24
    const unsigned short* aSrc  = A  + (long long)(mblk + arow) * K + ach;
    const unsigned short* bSrc0 = Bt + (long long)(nblk + arow) * K + ach;        // cols 0..63
    const unsigned short* bSrc1 = Bt + (long long)(nblk + arow + 64) * K + ach;   // cols 64..127

    unsigned aDst[2], bDst0[2], bDst1[2];
#pragma unroll
    for (int bf = 0; bf < 2; ++bf) {
        aDst[bf]  = (unsigned)(uintptr_t)&ldsA[bf][arow * 32 + ach];
        bDst0[bf] = (unsigned)(uintptr_t)&ldsB[bf][arow * 32 + ach];
        bDst1[bf] = (unsigned)(uintptr_t)&ldsB[bf][(arow + 64) * 32 + ach];
    }

    v8f acc[4];
#pragma unroll
    for (int t = 0; t < 4; ++t) acc[t] = (v8f){0.f, 0.f, 0.f, 0.f, 0.f, 0.f, 0.f, 0.f};

    int nk = K >> 5;
    // prologue: stage k=0 into buffer 0
    async_cp_b128(aDst[0], aSrc);
    async_cp_b128(bDst0[0], bSrc0);
    async_cp_b128(bDst1[0], bSrc1);

    for (int ki = 0; ki < nk; ++ki) {
        if (ki + 1 < nk) {
            int nb = (ki + 1) & 1;
            int k = (ki + 1) << 5;
            async_cp_b128(aDst[nb], aSrc + k);
            async_cp_b128(bDst0[nb], bSrc0 + k);
            async_cp_b128(bDst1[nb], bSrc1 + k);
            wait_async(3);          // current stage complete; next stage in flight
        } else {
            wait_async(0);
        }
        __syncthreads();            // publish LDS tile across all 8 waves

        const unsigned short* tA = ldsA[ki & 1];
        const unsigned short* tB = ldsB[ki & 1];
        v16bf a = load_a_lds(tA, wm);
#pragma unroll
        for (int t = 0; t < 4; ++t) {
            v16bf b = load_b_lds(tB, wn + 16 * t);
            acc[t] = WMMA_BF16(a, b, acc[t]);
        }
        __syncthreads();            // all waves done reading before refill
    }

    int lane = threadIdx.x & 31, l = lane & 15, hi = lane >> 4;
#pragma unroll
    for (int t = 0; t < 4; ++t) {
        int n = nblk + wn + 16 * t + l;
        float bn = bias[n];
#pragma unroll
        for (int i = 0; i < 8; ++i) {
            int m = mblk + wm + i + 8 * hi;
            float val = acc[t][i] + bn;
            if constexpr (EPI == EPI_Q) {
                int bb = m >> 11, ss = m & 2047, hh = n >> 6, dd = n & 63;
                outH[((((long long)bb * Hh + hh) * Sv + ss) << 6) + dd] = f2bf(val * 0.125f);
            } else if constexpr (EPI == EPI_K) {
                int bb = m >> 11, ss = m & 2047, hh = n >> 6, dd = n & 63;
                outH[((((long long)bb * Hh + hh) * Sv + ss) << 6) + dd] = f2bf(val);
            } else if constexpr (EPI == EPI_VT) {
                int bb = m >> 11, ss = m & 2047, hh = n >> 6, dd = n & 63;
                outH[(((long long)bb * Hh + hh) * HDv + dd) * Sv + ss] = f2bf(val);
            } else if constexpr (EPI == EPI_RESF32) {
                long long idx = (long long)m * N + n;
                outF[idx] = val + res[idx];
            } else { // EPI_GELU: quick_gelu = x * sigmoid(1.702 x)
                float ge = val / (1.f + __expf(-1.702f * val));
                outH[(long long)m * N + n] = f2bf(ge);
            }
        }
    }
}

// ---------------------------------------------------------------------------
// Flash attention (causal). q,k: bf16 [B*H, S, 64]; vt: bf16 [B*H, 64, S].
// out: bf16 [B, S, 1024]. One wave handles one 16-query tile.
// Computes S^T = K * Q^T so the score accumulator (query = lane-col, keys =
// acc rows) feeds the P-fragment of P*V with no cross-lane data movement.
// ---------------------------------------------------------------------------
__global__ __launch_bounds__(128) void attn_kernel(const unsigned short* __restrict__ q,
                                                   const unsigned short* __restrict__ k,
                                                   const unsigned short* __restrict__ vt,
                                                   unsigned short* __restrict__ o) {
    int bh = blockIdx.y;                 // 0..63
    int bb = bh >> 4, hh = bh & 15;
    int wv = threadIdx.x >> 5;
    int q0 = (blockIdx.x * 4 + wv) * 16; // query tile start

    const unsigned short* qh = q + (long long)bh * Sv * HDv;
    const unsigned short* kh = k + (long long)bh * Sv * HDv;
    const unsigned short* vh = vt + (long long)bh * HDv * Sv;

    int lane = threadIdx.x & 31, l = lane & 15, hi = lane >> 4;
    int qi = q0 + l;                     // this lane's query index (score layout)

    v16bf qf0 = load_b(qh, HDv, q0, 0);  // Q^T operand, head-dims 0..31
    v16bf qf1 = load_b(qh, HDv, q0, 32); // head-dims 32..63

    v8f oacc[4];
#pragma unroll
    for (int t = 0; t < 4; ++t) oacc[t] = (v8f){0.f, 0.f, 0.f, 0.f, 0.f, 0.f, 0.f, 0.f};
    float m_run = -1e30f, l_run = 0.f;

    for (int kt = 0; kt < q0 + 16; kt += 32) {
        // scores^T for two 16-key tiles
        v8f s0 = (v8f){0.f, 0.f, 0.f, 0.f, 0.f, 0.f, 0.f, 0.f};
        v8f s1 = s0;
        s0 = WMMA_BF16(load_a(kh, HDv, kt, 0), qf0, s0);
        s0 = WMMA_BF16(load_a(kh, HDv, kt, 32), qf1, s0);
        s1 = WMMA_BF16(load_a(kh, HDv, kt + 16, 0), qf0, s1);
        s1 = WMMA_BF16(load_a(kh, HDv, kt + 16, 32), qf1, s1);

        float sv[16];
        float tmax = -1e30f;
#pragma unroll
        for (int i = 0; i < 8; ++i) {
            int key0 = kt + i + 8 * hi;
            int key1 = key0 + 16;
            float a0 = (key0 <= qi) ? s0[i] : -1e30f;
            float a1 = (key1 <= qi) ? s1[i] : -1e30f;
            sv[i] = a0;
            sv[8 + i] = a1;
            tmax = fmaxf(tmax, fmaxf(a0, a1));
        }
        tmax = fmaxf(tmax, __shfl_xor(tmax, 16, 32));
        float m_new = fmaxf(m_run, tmax);
        float alpha = __expf(m_run - m_new);

        float lsum = 0.f;
        FragU pf;
#pragma unroll
        for (int e = 0; e < 16; ++e) {
            float p = __expf(sv[e] - m_new);
            lsum += p;
            pf.h[e] = f2bf(p);
        }
        lsum += __shfl_xor(lsum, 16, 32);
        l_run = l_run * alpha + lsum;
        m_run = m_new;

        // rescale O (rows are queries i + 8*hi; alpha lives at lane==query)
        float arow[8];
#pragma unroll
        for (int i = 0; i < 8; ++i) arow[i] = __shfl(alpha, i + 8 * hi, 32);
#pragma unroll
        for (int t = 0; t < 4; ++t)
#pragma unroll
            for (int i = 0; i < 8; ++i) oacc[t][i] *= arow[i];

        // O += P * V  (V as B-fragment from head-dim-major storage)
        v16bf pv = pf.v;
#pragma unroll
        for (int t = 0; t < 4; ++t)
            oacc[t] = WMMA_BF16(pv, load_b(vh, Sv, t * 16, kt), oacc[t]);
    }

    float linv = 1.f / l_run;
    float lrow[8];
#pragma unroll
    for (int i = 0; i < 8; ++i) lrow[i] = __shfl(linv, i + 8 * hi, 32);

#pragma unroll
    for (int t = 0; t < 4; ++t) {
        int col = hh * HDv + t * 16 + l;
#pragma unroll
        for (int i = 0; i < 8; ++i) {
            int row = q0 + i + 8 * hi;
            o[((long long)(bb * Sv + row)) * Dv + col] = f2bf(oacc[t][i] * lrow[i]);
        }
    }
}

// ---------------------------------------------------------------------------
// Host-side orchestration
// ---------------------------------------------------------------------------
extern "C" void kernel_launch(void* const* d_in, const int* in_sizes, int n_in,
                              void* d_out, int out_size, void* d_ws, size_t ws_size,
                              hipStream_t stream) {
    const float* x     = (const float*)d_in[0];
    const float* ln1_g = (const float*)d_in[1];
    const float* ln1_b = (const float*)d_in[2];
    const float* Wq    = (const float*)d_in[3];
    const float* bq    = (const float*)d_in[4];
    const float* Wk    = (const float*)d_in[5];
    const float* bk    = (const float*)d_in[6];
    const float* Wv    = (const float*)d_in[7];
    const float* bv    = (const float*)d_in[8];
    const float* Wo    = (const float*)d_in[9];
    const float* bo    = (const float*)d_in[10];
    const float* ln2_g = (const float*)d_in[11];
    const float* ln2_b = (const float*)d_in[12];
    const float* W1    = (const float*)d_in[13];
    const float* b1    = (const float*)d_in[14];
    const float* W2    = (const float*)d_in[15];
    const float* b2    = (const float*)d_in[16];

    unsigned char* w = (unsigned char*)d_ws;
    size_t off = 0;
    auto take = [&](size_t bytes) { void* p = w + off; off += (bytes + 255) & ~size_t(255); return p; };

    unsigned short* WqT = (unsigned short*)take((size_t)Dv * Dv * 2);
    unsigned short* WkT = (unsigned short*)take((size_t)Dv * Dv * 2);
    unsigned short* WvT = (unsigned short*)take((size_t)Dv * Dv * 2);
    unsigned short* WoT = (unsigned short*)take((size_t)Dv * Dv * 2);
    unsigned short* W1T = (unsigned short*)take((size_t)Dv * DFF * 2);
    unsigned short* W2T = (unsigned short*)take((size_t)DFF * Dv * 2);
    unsigned short* hln = (unsigned short*)take((size_t)ROWS * Dv * 2);  // ln1 out, later attn out
    unsigned short* qb  = (unsigned short*)take((size_t)ROWS * Dv * 2);  // q, later ln2 out
    unsigned short* kb  = (unsigned short*)take((size_t)ROWS * Dv * 2);
    unsigned short* vtb = (unsigned short*)take((size_t)ROWS * Dv * 2);
    float*          x2  = (float*)take((size_t)ROWS * Dv * 4);
    unsigned short* mid = (unsigned short*)take((size_t)ROWS * DFF * 2);
    (void)ws_size;

    // 1) weights -> bf16 transposed [N,K]
    wt_convert<<<(Dv * Dv) / 256, 256, 0, stream>>>(Wq, WqT, Dv, Dv);
    wt_convert<<<(Dv * Dv) / 256, 256, 0, stream>>>(Wk, WkT, Dv, Dv);
    wt_convert<<<(Dv * Dv) / 256, 256, 0, stream>>>(Wv, WvT, Dv, Dv);
    wt_convert<<<(Dv * Dv) / 256, 256, 0, stream>>>(Wo, WoT, Dv, Dv);
    wt_convert<<<(Dv * DFF) / 256, 256, 0, stream>>>(W1, W1T, Dv, DFF);
    wt_convert<<<(DFF * Dv) / 256, 256, 0, stream>>>(W2, W2T, DFF, Dv);

    // 2) LN1
    ln_kernel<<<ROWS, 256, 0, stream>>>(x, ln1_g, ln1_b, hln);

    // 3) QKV projections
    dim3 g1(ROWS / 64, Dv / 128), b256(256);
    gemm_k<EPI_Q><<<g1, b256, 0, stream>>>(hln, WqT, bq, nullptr, nullptr, qb, ROWS, Dv, Dv);
    gemm_k<EPI_K><<<g1, b256, 0, stream>>>(hln, WkT, bk, nullptr, nullptr, kb, ROWS, Dv, Dv);
    gemm_k<EPI_VT><<<g1, b256, 0, stream>>>(hln, WvT, bv, nullptr, nullptr, vtb, ROWS, Dv, Dv);

    // 4) causal flash attention -> hln (reuse)
    attn_kernel<<<dim3(Sv / 16 / 4, Bv * Hh), 128, 0, stream>>>(qb, kb, vtb, hln);

    // 5) out-proj + residual -> x2 (fp32)
    gemm_k<EPI_RESF32><<<g1, b256, 0, stream>>>(hln, WoT, bo, x, x2, nullptr, ROWS, Dv, Dv);

    // 6) LN2 -> qb (reuse)
    ln_kernel<<<ROWS, 256, 0, stream>>>(x2, ln2_g, ln2_b, qb);

    // 7) MLP up + quickGELU -> mid (bf16)
    dim3 g2(ROWS / 64, DFF / 128);
    gemm_k<EPI_GELU><<<g2, b256, 0, stream>>>(qb, W1T, b1, nullptr, nullptr, mid, ROWS, DFF, Dv);

    // 8) MLP down + residual -> d_out (fp32)
    gemm_k<EPI_RESF32><<<g1, b256, 0, stream>>>(mid, W2T, b2, x2, (float*)d_out, nullptr, ROWS, Dv, DFF);
}
